// ChebNet_90288802496749
// MI455X (gfx1250) — matrix operand order repaired and checked by reading, not compiled, not run on hardware
//
#include <hip/hip_runtime.h>

// ---------------------------------------------------------------------------
// Problem constants (from reference)
// ---------------------------------------------------------------------------
#define NGRAPH   2048
#define NPG      75          // nodes per graph
#define DEG      8
#define EPG      (NPG * DEG) // 600 edges per graph
#define NNODES   (NGRAPH * NPG)   // 153600
#define NEDGES   (NGRAPH * EPG)   // 1228800
#define CHEB_K   5

// ---------------------------------------------------------------------------
// CDNA5 WMMA / vector types
// ---------------------------------------------------------------------------
typedef __attribute__((ext_vector_type(16))) __bf16 v16bf;
typedef __attribute__((ext_vector_type(8)))  __bf16 v8bf;
typedef __attribute__((ext_vector_type(8)))  float  v8f;
typedef __attribute__((ext_vector_type(4)))  unsigned u32x4;
typedef __attribute__((ext_vector_type(8)))  unsigned u32x8;

static __device__ inline unsigned bfbits(float f) {
    unsigned u = __builtin_bit_cast(unsigned, f);
    return ((u + 0x7FFFu + ((u >> 16) & 1u)) >> 16) & 0xFFFFu;  // RNE
}
static __device__ inline unsigned pack2bf(float lo, float hi) {
    return (bfbits(hi) << 16) | bfbits(lo);
}
static __device__ inline v8f zero8() {
    v8f z = {0.f, 0.f, 0.f, 0.f, 0.f, 0.f, 0.f, 0.f};
    return z;
}
// LDS byte offset of a generic pointer to __shared__ (low 32 bits of flat addr)
static __device__ inline unsigned lds_off(const void* p) {
    return (unsigned)(size_t)p;
}

// ---------------------------------------------------------------------------
// Tensor Data Mover: 2D tile load (32 cols x 75 rows of f32, row stride FIN)
// into LDS.  D# per ISA 8.3/8.4 (2-group form, tensors up to 2D).
// Issued wave-uniform; tracked with TENSORcnt.
// ---------------------------------------------------------------------------
template<int FIN>
static __device__ inline void tdm_load_slice(const float* gsrc, unsigned lds_addr)
{
    unsigned long long ga = (unsigned long long)(size_t)gsrc;
    u32x4 g0;
    g0[0] = 1u;                                            // count=1 (valid user D#)
    g0[1] = lds_addr;                                      // lds_addr
    g0[2] = (unsigned)(ga & 0xFFFFFFFFu);                  // global_addr[31:0]
    g0[3] = (unsigned)((ga >> 32) & 0x01FFFFFFu)           // global_addr[56:32]
          | (2u << 30);                                    // type = 2 ("image")
    u32x8 g1;
    g1[0] = 2u << 16;                                      // data_size = 4B
    g1[1] = ((unsigned)FIN & 0xFFFFu) << 16;               // tensor_dim0[15:0]
    g1[2] = ((unsigned)FIN >> 16) | ((unsigned)NPG << 16); // dim0[31:16] | dim1[15:0]
    g1[3] = (32u << 16);                                   // dim1[31:16]=0 | tile_dim0=32
    g1[4] = (unsigned)NPG;                                 // tile_dim1=75 | tile_dim2=0
    g1[5] = (unsigned)FIN;                                 // tensor_dim0_stride[31:0]
    g1[6] = 0u;                                            // stride[47:32] | dim1_stride lo
    g1[7] = 0u;
    asm volatile("tensor_load_to_lds %0, %1" :: "s"(g0), "s"(g1) : "memory");
}

// ---------------------------------------------------------------------------
// Kernel 1: per-graph Laplacian preprocessing
// ---------------------------------------------------------------------------
__global__ __launch_bounds__(128) void preprocess_kernel(
    const int*   __restrict__ ei,       // [2][E]
    const float* __restrict__ lmax,     // [G]
    float*          __restrict__ w_edge, // [E]
    unsigned short* __restrict__ rc,     // [E] packed local (src | dst<<8)
    float*          __restrict__ diag)   // [N]
{
    const int g   = blockIdx.x;
    const int tid = threadIdx.x;
    __shared__ float deg[80];
    __shared__ float dinv[80];

    for (int i = tid; i < 80; i += 128) deg[i] = 0.0f;
    __syncthreads();

    const int eb = g * EPG;
    for (int e = tid; e < EPG; e += 128) {
        int r = ei[eb + e]          - g * NPG;
        int c = ei[NEDGES + eb + e] - g * NPG;
        float ew = (r == c) ? 0.0f : 1.0f;
        atomicAdd(&deg[r], ew);
    }
    __syncthreads();

    const float li = 2.0f / lmax[g];
    for (int n = tid; n < NPG; n += 128) {
        float d = deg[n];
        dinv[n] = (d > 0.0f) ? rsqrtf(d) : 0.0f;
        diag[g * NPG + n] = li - 1.0f;
    }
    __syncthreads();

    for (int e = tid; e < EPG; e += 128) {
        int r = ei[eb + e]          - g * NPG;
        int c = ei[NEDGES + eb + e] - g * NPG;
        float ew = (r == c) ? 0.0f : 1.0f;
        w_edge[eb + e] = -li * dinv[r] * ew * dinv[c];
        rc[eb + e] = (unsigned short)(r | (c << 8));
    }
}

// ---------------------------------------------------------------------------
// Kernel 1b: pre-pack weights into bf16 B fragments (contiguous k = 16*(lane/16) + i)
// ---------------------------------------------------------------------------
template<int FIN, int FOUT>
__global__ __launch_bounds__(256) void pack_weights_kernel(
    const float* __restrict__ W, unsigned* __restrict__ Wp /* bf16 pairs */)
{
    constexpr int NT  = FOUT / 16;
    constexpr int NSL = (FIN + 31) / 32;
    const int total = CHEB_K * NSL * NT * 32;
    int idx = blockIdx.x * 256 + threadIdx.x;     // (k,s,nt,lane)
    if (idx >= total) return;
    const int lane = idx & 31;
    int t = idx >> 5;
    const int nt = t % NT;  t /= NT;
    const int s  = t % NSL;
    const int k  = t / NSL;
    const int n  = lane & 15;
    const int kg = (lane >> 4) * 16;

    unsigned pk[8];
#pragma unroll
    for (int i = 0; i < 8; ++i) {
        int r0 = s * 32 + kg + 2 * i;
        float w0 = (r0     < FIN) ? W[((size_t)k * FIN + r0    ) * FOUT + nt * 16 + n] : 0.0f;
        float w1 = (r0 + 1 < FIN) ? W[((size_t)k * FIN + r0 + 1) * FOUT + nt * 16 + n] : 0.0f;
        pk[i] = pack2bf(w0, w1);
    }
    unsigned* dst = Wp + (size_t)idx * 8;         // 32 bytes per lane-fragment
#pragma unroll
    for (int i = 0; i < 8; ++i) dst[i] = pk[i];
}

// ---------------------------------------------------------------------------
// lhat step in LDS:  out = diag*in + scatter(w_e * in[src] -> dst)
//                    if prev: out = 2*out - prev  (Chebyshev recursion)
// Finalize pass also packs the result into the bf16 A-staging buffer.
// ---------------------------------------------------------------------------
static __device__ void lhat_step(float (*out)[32], const float (*in)[32],
                                 const float (*prev)[32], __bf16 (*abf)[32],
                                 const float* sw, const unsigned char* ssrc,
                                 const unsigned char* sdst, const float* sdiag,
                                 int tid)
{
    __syncthreads();   // prior WMMA reads of out/abf buffers must be done
    for (int idx = tid; idx < 80 * 8; idx += 256) {
        int n = idx >> 3, q = idx & 7;
        float4 v = ((const float4*)in[n])[q];
        const float d = sdiag[n];
        v.x *= d; v.y *= d; v.z *= d; v.w *= d;
        ((float4*)out[n])[q] = v;
    }
    __syncthreads();
    // edge-major scatter: per-edge header loaded once, source row streamed b128
    for (int e = tid; e < EPG; e += 256) {
        const float w  = sw[e];
        const int   rs = ssrc[e];
        const int   rd = sdst[e];
        const float4* vin = (const float4*)in[rs];
        float* vout = out[rd];
#pragma unroll
        for (int q = 0; q < 8; ++q) {
            float4 v = vin[q];
            atomicAdd(&vout[4 * q + 0], w * v.x);   // ds_add_f32
            atomicAdd(&vout[4 * q + 1], w * v.y);
            atomicAdd(&vout[4 * q + 2], w * v.z);
            atomicAdd(&vout[4 * q + 3], w * v.w);
        }
    }
    __syncthreads();
    // finalize (+ optional recursion combine) and bf16 staging, pair-packed
    if (prev) {
        for (int idx = tid; idx < 80 * 8; idx += 256) {
            int n = idx >> 3, q = idx & 7;
            float4 o = ((const float4*)out[n])[q];
            float4 p = ((const float4*)prev[n])[q];
            o.x = 2.0f * o.x - p.x;  o.y = 2.0f * o.y - p.y;
            o.z = 2.0f * o.z - p.z;  o.w = 2.0f * o.w - p.w;
            ((float4*)out[n])[q] = o;
            uint2 pk; pk.x = pack2bf(o.x, o.y); pk.y = pack2bf(o.z, o.w);
            *(uint2*)&abf[n][q * 4] = pk;
        }
    } else {
        for (int idx = tid; idx < 80 * 8; idx += 256) {
            int n = idx >> 3, q = idx & 7;
            float4 o = ((const float4*)out[n])[q];
            uint2 pk; pk.x = pack2bf(o.x, o.y); pk.y = pack2bf(o.z, o.w);
            *(uint2*)&abf[n][q * 4] = pk;
        }
    }
    __syncthreads();
}

// ---------------------------------------------------------------------------
// One WMMA accumulation step: acc[mt] += A(bf16 LDS staging) x B(packed Wp).
// ---------------------------------------------------------------------------
template<int FIN, int FOUT>
static __device__ inline void wmma_step(v8f* acc, const __bf16 (*abf)[32],
                                        const unsigned* __restrict__ Wp,
                                        int k, int s, int wv, int lane)
{
    constexpr int NT  = FOUT / 16;
    constexpr int NSL = (FIN + 31) / 32;
    if (wv >= NT) return;                 // wave-uniform

    const v8bf* wp = (const v8bf*)(Wp + ((((size_t)k * NSL + s) * NT + wv) * 32 + lane) * 8);
    v8bf blo = wp[0];
    v8bf bhi = wp[1];
    v16bf bf = __builtin_shufflevector(blo, bhi,
        0, 1, 2, 3, 4, 5, 6, 7, 8, 9, 10, 11, 12, 13, 14, 15);

    const int m0 = lane & 15;
    const int ag = (lane >> 4) * 16;      // lane half selects K group
#pragma unroll
    for (int mt = 0; mt < 5; ++mt) {
        const int m = mt * 16 + m0;
        v8bf alo = *(const v8bf*)&abf[m][ag];
        v8bf ahi = *(const v8bf*)&abf[m][ag + 8];
        v16bf af = __builtin_shufflevector(alo, ahi,
            0, 1, 2, 3, 4, 5, 6, 7, 8, 9, 10, 11, 12, 13, 14, 15);
        acc[mt] = __builtin_amdgcn_wmma_f32_16x16x32_bf16(
            false, af, false, bf, (short)0, acc[mt], false, false);
    }
}

// ---------------------------------------------------------------------------
// Kernel 2: fused ChebConv layer. One workgroup (8 wave32) per graph.
// Edge tables arrive via GLOBAL_LOAD_ASYNC_TO_LDS (ASYNCcnt); feature slices
// arrive via the Tensor Data Mover (TENSORcnt), double-buffered so the DMA of
// slice s+1 overlaps the whole 5-term recursion of slice s.
// ---------------------------------------------------------------------------
template<int FIN, int FOUT>
__global__ __launch_bounds__(256, 1) void cheb_conv_kernel(
    const float* __restrict__ hin,      // [N, FIN]
    float*       __restrict__ hout,     // [N, FOUT]
    const unsigned* __restrict__ Wp,    // packed bf16 weights
    const float* __restrict__ bias,     // [FOUT]
    const float* __restrict__ w_edge,   // [E]
    const unsigned short* __restrict__ rc, // [E]
    const float* __restrict__ diag)     // [N]
{
    constexpr int NT  = FOUT / 16;            // waves doing WMMA
    constexpr int NSL = (FIN + 31) / 32;      // 32-wide input-feature slices
    const int g    = blockIdx.x;
    const int tid  = threadIdx.x;
    const int lane = tid & 31;
    const int wv   = tid >> 5;

    __shared__ alignas(16) float t0[80][32];
    __shared__ alignas(16) float t1[80][32];
    __shared__ alignas(16) float t2[80][32];
    __shared__ alignas(16) float tin[80][32];    // TDM landing buffer
    __shared__ alignas(16) __bf16 abf[80][32];   // A-fragment staging (bf16)
    __shared__ alignas(16) float sw[EPG];
    __shared__ alignas(16) unsigned short rcs[EPG];
    __shared__ unsigned char ssrc[EPG], sdst[EPG];
    __shared__ float sdiag[80];

    __builtin_prefetch(Wp + (size_t)tid * 8, 0, 3);   // global_prefetch_b8

    // ---- async edge-table copies (ASYNCcnt), overlap with everything below
    {
        const char* wb = (const char*)(w_edge + (size_t)g * EPG);
        for (int t = tid; t < EPG / 4; t += 256) {        // 150 x b128
            unsigned l  = lds_off(&sw[t * 4]);
            unsigned go = (unsigned)(t * 16);
            asm volatile("global_load_async_to_lds_b128 %0, %1, %2 offset:0"
                         :: "v"(l), "v"(go), "s"(wb) : "memory");
        }
        const char* rb = (const char*)(rc + (size_t)g * EPG);
        for (int t = tid; t < EPG / 8; t += 256) {        // 75 x b128
            unsigned l  = lds_off(&rcs[t * 8]);
            unsigned go = (unsigned)(t * 16);
            asm volatile("global_load_async_to_lds_b128 %0, %1, %2 offset:0"
                         :: "v"(l), "v"(go), "s"(rb) : "memory");
        }
    }
    // ---- first feature tile via TDM
    if constexpr (FIN >= 32) {
        if (wv == 0)
            tdm_load_slice<FIN>(hin + (size_t)g * NPG * FIN, lds_off(tin));
    }
    for (int n = tid; n < 80; n += 256)
        sdiag[n] = (n < NPG) ? diag[g * NPG + n] : 0.0f;

    asm volatile("s_wait_asynccnt 0x0" ::: "memory");
    __syncthreads();
    for (int e = tid; e < EPG; e += 256) {
        unsigned short p = rcs[e];
        ssrc[e] = (unsigned char)(p & 0xFF);
        sdst[e] = (unsigned char)(p >> 8);
    }

    v8f acc[5];
#pragma unroll
    for (int i = 0; i < 5; ++i) acc[i] = zero8();

    for (int s = 0; s < NSL; ++s) {
        if constexpr (FIN >= 32) {
            if (wv == 0) __builtin_amdgcn_s_wait_tensorcnt(0);  // tile s landed
        }
        __syncthreads();     // tile visible to all; buffers free to overwrite
        if constexpr (FIN >= 32) {
            // T0 = staged tile (rows 75..79 zero) + bf16 staging
            for (int idx = tid; idx < 80 * 8; idx += 256) {
                int n = idx >> 3, q = idx & 7;
                float4 v; v.x = v.y = v.z = v.w = 0.0f;
                if (n < NPG) v = ((const float4*)tin[n])[q];
                ((float4*)t0[n])[q] = v;
                uint2 pk; pk.x = pack2bf(v.x, v.y); pk.y = pack2bf(v.z, v.w);
                *(uint2*)&abf[n][q * 4] = pk;
            }
            __syncthreads();                        // tin fully consumed
            if (wv == 0 && s + 1 < NSL)             // prefetch next tile
                tdm_load_slice<FIN>(hin + (size_t)g * NPG * FIN + (s + 1) * 32,
                                    lds_off(tin));
        } else {
            // FIN==2: direct path (tiny, zero-padded to 32 cols)
            for (int idx = tid; idx < 80 * 16; idx += 256) {
                int n = idx >> 4, f = (idx & 15) * 2;
                float v0 = 0.0f, v1 = 0.0f;
                if (n < NPG) {
                    const float* row = hin + (size_t)(g * NPG + n) * FIN + f;
                    if (f     < FIN) v0 = row[0];
                    if (f + 1 < FIN) v1 = row[1];
                }
                t0[n][f] = v0; t0[n][f + 1] = v1;
                *(unsigned*)&abf[n][f] = pack2bf(v0, v1);
            }
            __syncthreads();
        }

        wmma_step<FIN, FOUT>(acc, abf, Wp, 0, s, wv, lane);            // k = 0
        lhat_step(t1, t0, nullptr, abf, sw, ssrc, sdst, sdiag, tid);   // T1 = L^ T0
        wmma_step<FIN, FOUT>(acc, abf, Wp, 1, s, wv, lane);            // k = 1

        float (*p0)[32] = t0; float (*p1)[32] = t1; float (*p2)[32] = t2;
        for (int k = 2; k < CHEB_K; ++k) {
            lhat_step(p2, p1, p0, abf, sw, ssrc, sdst, sdiag, tid);    // Tk = 2 L^ Tk-1 - Tk-2
            wmma_step<FIN, FOUT>(acc, abf, Wp, k, s, wv, lane);
            float (*tmp)[32] = p0; p0 = p1; p1 = p2; p2 = tmp;
        }
    }

    // epilogue: bias + ReLU + store (C/D fragment layout, ISA 7.12.2)
    if (wv < NT) {
        const int n   = lane & 15;
        const int col = wv * 16 + n;
        const float b = bias[col];
        const int rbase = (lane >> 4) * 8;
#pragma unroll
        for (int mt = 0; mt < 5; ++mt) {
#pragma unroll
            for (int i = 0; i < 8; ++i) {
                int row = mt * 16 + rbase + i;
                if (row < NPG) {
                    float v = acc[mt][i] + b;
                    hout[(size_t)(g * NPG + row) * FOUT + col] = v > 0.0f ? v : 0.0f;
                }
            }
        }
    }
}

// ---------------------------------------------------------------------------
// Kernel 3: mean-pool per graph + fc1(ReLU) + fc2 + log_softmax
// ---------------------------------------------------------------------------
__global__ __launch_bounds__(128) void head_kernel(
    const float* __restrict__ h3,     // [N,128]
    const float* __restrict__ fc1_w,  // [128,32]
    const float* __restrict__ fc1_b,  // [32]
    const float* __restrict__ fc2_w,  // [32,10]
    const float* __restrict__ fc2_b,  // [10]
    float*       __restrict__ out)    // [G,10]
{
    const int g = blockIdx.x, tid = threadIdx.x;
    __shared__ float pooled[128];
    __shared__ float a1[32];
    __shared__ float logits[10];

    float s = 0.0f;
    for (int n = 0; n < NPG; ++n)
        s += h3[((size_t)g * NPG + n) * 128 + tid];
    pooled[tid] = s * (1.0f / (float)NPG);
    __syncthreads();

    if (tid < 32) {
        float acc = fc1_b[tid];
        for (int f = 0; f < 128; ++f) acc += pooled[f] * fc1_w[f * 32 + tid];
        a1[tid] = acc > 0.0f ? acc : 0.0f;
    }
    __syncthreads();

    if (tid < 10) {
        float acc = fc2_b[tid];
        for (int i = 0; i < 32; ++i) acc += a1[i] * fc2_w[i * 10 + tid];
        logits[tid] = acc;
    }
    __syncthreads();

    if (tid < 10) {
        float m = logits[0];
        for (int j = 1; j < 10; ++j) m = fmaxf(m, logits[j]);
        float sum = 0.0f;
        for (int j = 0; j < 10; ++j) sum += expf(logits[j] - m);
        out[g * 10 + tid] = logits[tid] - m - logf(sum);
    }
}

// ---------------------------------------------------------------------------
// Launch
// ---------------------------------------------------------------------------
extern "C" void kernel_launch(void* const* d_in, const int* in_sizes, int n_in,
                              void* d_out, int out_size, void* d_ws, size_t ws_size,
                              hipStream_t stream)
{
    (void)in_sizes; (void)n_in; (void)out_size; (void)ws_size;

    const float* x    = (const float*)d_in[0];
    const int*   ei   = (const int*)d_in[1];
    // d_in[2] = batch (implicit: node/NPG), unused
    const float* lmax = (const float*)d_in[3];
    const float* W1   = (const float*)d_in[4];
    const float* b1   = (const float*)d_in[5];
    const float* W2   = (const float*)d_in[6];
    const float* b2   = (const float*)d_in[7];
    const float* W3   = (const float*)d_in[8];
    const float* b3   = (const float*)d_in[9];
    const float* f1w  = (const float*)d_in[10];
    const float* f1b  = (const float*)d_in[11];
    const float* f2w  = (const float*)d_in[12];
    const float* f2b  = (const float*)d_in[13];

    char* ws = (char*)d_ws;
    float* w_edge       = (float*)ws;          ws += (size_t)NEDGES * 4;
    unsigned short* rc  = (unsigned short*)ws; ws += (size_t)NEDGES * 2;
    float* diag         = (float*)ws;          ws += (size_t)NNODES * 4;
    float* bufA         = (float*)ws;          ws += (size_t)NNODES * 128 * 4;
    float* bufB         = (float*)ws;          ws += (size_t)NNODES * 128 * 4;
    // packed bf16 weight fragments (sized: CHEB_K * NSL * NT * 32 lanes * 32B)
    unsigned* Wp1       = (unsigned*)ws;       ws += (size_t)CHEB_K * 1 * 4 * 32 * 32;
    unsigned* Wp2       = (unsigned*)ws;       ws += (size_t)CHEB_K * 2 * 8 * 32 * 32;
    unsigned* Wp3       = (unsigned*)ws;       ws += (size_t)CHEB_K * 4 * 8 * 32 * 32;

    preprocess_kernel<<<NGRAPH, 128, 0, stream>>>(ei, lmax, w_edge, rc, diag);
    pack_weights_kernel<2,   64><<<(CHEB_K * 1 * 4 * 32 + 255) / 256, 256, 0, stream>>>(W1, Wp1);
    pack_weights_kernel<64, 128><<<(CHEB_K * 2 * 8 * 32 + 255) / 256, 256, 0, stream>>>(W2, Wp2);
    pack_weights_kernel<128,128><<<(CHEB_K * 4 * 8 * 32 + 255) / 256, 256, 0, stream>>>(W3, Wp3);

    cheb_conv_kernel<2,   64><<<NGRAPH, 256, 0, stream>>>(x,    bufA, Wp1, b1, w_edge, rc, diag);
    cheb_conv_kernel<64, 128><<<NGRAPH, 256, 0, stream>>>(bufA, bufB, Wp2, b2, w_edge, rc, diag);
    cheb_conv_kernel<128,128><<<NGRAPH, 256, 0, stream>>>(bufB, bufA, Wp3, b3, w_edge, rc, diag);
    head_kernel<<<NGRAPH, 128, 0, stream>>>(bufA, f1w, f1b, f2w, f2b, (float*)d_out);
}
